// LSTMTagger_78752520339503
// MI455X (gfx1250) — matrix operand order repaired and checked by reading, not compiled, not run on hardware
//
#include <hip/hip_runtime.h>
#include <hip/hip_bf16.h>

// ---------------- types ----------------
typedef __attribute__((ext_vector_type(16))) __bf16 v16bf;
typedef __attribute__((ext_vector_type(8)))  float  v8f;
typedef __attribute__((ext_vector_type(8)))  __bf16 v8bf;

union FragB { v16bf v; uint4 q[2]; };

#define E_   256
#define HC_  256
#define HW_  512
#define T_   64
#define S_   4096
#define L_   16

__device__ __forceinline__ float sigf(float x)     { return 1.0f / (1.0f + __expf(-x)); }
__device__ __forceinline__ float tanhfast(float x) { return 1.0f - 2.0f / (__expf(2.0f * x) + 1.0f); }

// ---------------- prep: f32 -> bf16 ----------------
__global__ void cvt_bf16(const float* __restrict__ src, __bf16* __restrict__ dst, int n) {
    int i = blockIdx.x * blockDim.x + threadIdx.x;
    if (i < n) dst[i] = (__bf16)src[i];
}

// gather word embeddings into xw[:, 0:256] (bf16)
__global__ void gather_we(const int* __restrict__ sentence,
                          const float* __restrict__ word_emb,
                          __bf16* __restrict__ xw) {
    int s = blockIdx.x;
    int j = threadIdx.x;                       // 0..255
    long row = (long)sentence[s] * E_;
    xw[(long)s * (E_ + HC_) + j] = (__bf16)word_emb[row + j];
}

// ---------------- phase 1: char LSTM (WMMA bf16) ----------------
// 32 words / block, 512 threads = 16 waves. Wave w: m-tile = w&1, j-group = w>>1
// (32 hidden cols), owning the i/f/g/o tiles for those cols -> cell update in regs.
__global__ __launch_bounds__(512)
void char_lstm(const int* __restrict__ words_chars,   // 4096 x 16
               const int* __restrict__ word_lens,     // 4096
               const __bf16* __restrict__ ce,         // 26 x 256
               const __bf16* __restrict__ Wih,        // 1024 x 256 (row n = K-contig)
               const __bf16* __restrict__ Whh,        // 1024 x 256
               const float* __restrict__ bc,          // 1024
               __bf16* __restrict__ xw)               // 4096 x 512, write cols 256..511
{
    __shared__ __bf16 lx[32 * 256];   // x_t tile
    __shared__ __bf16 lh[32 * 256];   // h state tile
    __shared__ int    lwl[32];

    const int tid   = threadIdx.x;
    const int lane  = tid & 31;
    const int wv    = tid >> 5;
    const int mt    = wv & 1;          // m-tile (16 words)
    const int jg    = wv >> 1;         // j-group: cols [jg*32, jg*32+32)
    const int l15   = lane & 15;
    const int hi    = lane >> 4;
    const int sbase = blockIdx.x * 32;

    if (tid < 32) lwl[tid] = word_lens[sbase + tid];
    {   // zero h state: 16KB / 512 threads = 32B each
        uint4 z4 = make_uint4(0u, 0u, 0u, 0u);
        uint4* p = (uint4*)lh;
        p[tid * 2 + 0] = z4;
        p[tid * 2 + 1] = z4;
    }

    v8f cfr[2];                        // c state, 2 j-tiles x 8 rows, in registers
    #pragma unroll
    for (int jt = 0; jt < 2; jt++)
        #pragma unroll
        for (int r = 0; r < 8; r++) cfr[jt][r] = 0.0f;

    float bias[8];                     // per-acc-tile bias (per-N, constant over M)
    #pragma unroll
    for (int g = 0; g < 4; g++)
        #pragma unroll
        for (int jt = 0; jt < 2; jt++)
            bias[g * 2 + jt] = bc[g * 256 + jg * 32 + jt * 16 + l15];

    const int arow = mt * 16 + l15;    // A-operand row for this lane

    for (int t = 0; t < L_; t++) {
        // ---- gather x_t tile: 32 words x 256 bf16, 16 elems (32B) per thread
        {
            int wd = tid >> 4, ch = tid & 15;
            int ci = words_chars[(sbase + wd) * L_ + t];
            const uint4* src = (const uint4*)(ce + ci * E_ + ch * 16);
            uint4*       dst = (uint4*)(lx + wd * 256 + ch * 16);
            dst[0] = src[0];
            dst[1] = src[1];
        }
        __syncthreads();   // x visible; prev-iter h writes visible

        v8f acc[8];
        #pragma unroll
        for (int i = 0; i < 8; i++)
            #pragma unroll
            for (int r = 0; r < 8; r++) acc[i][r] = bias[i];

        // ---- z += x_t @ Wih^T  (K = 256)
        #pragma unroll 2
        for (int kt = 0; kt < 8; kt++) {
            FragB a;
            int k0 = kt * 32 + hi * 8;
            a.q[0] = *(const uint4*)(lx + arow * 256 + k0);
            a.q[1] = *(const uint4*)(lx + arow * 256 + k0 + 16);
            int kb = kt * 32 + hi * 8;
            #pragma unroll
            for (int g = 0; g < 4; g++)
                #pragma unroll
                for (int jt = 0; jt < 2; jt++) {
                    int n = g * 256 + jg * 32 + jt * 16 + l15;
                    FragB b;
                    const uint4* wp = (const uint4*)(Wih + n * 256 + kb);
                    b.q[0] = wp[0];
                    b.q[1] = wp[2];
                    acc[g * 2 + jt] = __builtin_amdgcn_wmma_f32_16x16x32_bf16(
                        false, a.v, false, b.v, (short)0, acc[g * 2 + jt], false, false);
                }
        }
        // ---- z += h @ Whh^T  (K = 256)
        #pragma unroll 2
        for (int kt = 0; kt < 8; kt++) {
            FragB a;
            int k0 = kt * 32 + hi * 8;
            a.q[0] = *(const uint4*)(lh + arow * 256 + k0);
            a.q[1] = *(const uint4*)(lh + arow * 256 + k0 + 16);
            int kb = kt * 32 + hi * 8;
            #pragma unroll
            for (int g = 0; g < 4; g++)
                #pragma unroll
                for (int jt = 0; jt < 2; jt++) {
                    int n = g * 256 + jg * 32 + jt * 16 + l15;
                    FragB b;
                    const uint4* wp = (const uint4*)(Whh + n * 256 + kb);
                    b.q[0] = wp[0];
                    b.q[1] = wp[2];
                    acc[g * 2 + jt] = __builtin_amdgcn_wmma_f32_16x16x32_bf16(
                        false, a.v, false, b.v, (short)0, acc[g * 2 + jt], false, false);
                }
        }

        // ---- LSTM cell update: i/f/g/o for (word m, col j) all in this lane
        float hnew[2][8];
        #pragma unroll
        for (int jt = 0; jt < 2; jt++) {
            #pragma unroll
            for (int r = 0; r < 8; r++) {
                int m = mt * 16 + hi * 8 + r;
                bool act = t < lwl[m];
                float iv = acc[0 + jt][r];
                float fv = acc[2 + jt][r];
                float gv = acc[4 + jt][r];
                float ov = acc[6 + jt][r];
                float cn = sigf(fv) * cfr[jt][r] + sigf(iv) * tanhfast(gv);
                float hn = sigf(ov) * tanhfast(cn);
                int j = jg * 32 + jt * 16 + l15;
                float hold = (float)lh[m * 256 + j];
                cfr[jt][r]  = act ? cn : cfr[jt][r];
                hnew[jt][r] = act ? hn : hold;
            }
        }
        __syncthreads();   // all waves done reading lh/lx
        #pragma unroll
        for (int jt = 0; jt < 2; jt++)
            #pragma unroll
            for (int r = 0; r < 8; r++) {
                int m = mt * 16 + hi * 8 + r;
                int j = jg * 32 + jt * 16 + l15;
                lh[m * 256 + j] = (__bf16)hnew[jt][r];
            }
        // next-iter top sync orders these writes before the next GEMM
    }
    __syncthreads();
    // ---- write final h -> xw[:, 256:512]
    {
        int wd = tid >> 4, ch = tid & 15;
        const uint4* src = (const uint4*)(lh + wd * 256 + ch * 16);
        uint4*       dst = (uint4*)(xw + (long)(sbase + wd) * 512 + 256 + ch * 16);
        dst[0] = src[0];
        dst[1] = src[1];
    }
}

// ---------------- phase 2: word LSTM (sequential, 1 persistent workgroup) ----------------
__global__ __launch_bounds__(1024)
void word_lstm(const __bf16* __restrict__ xw,    // 4096 x 512 inputs
               const __bf16* __restrict__ Wih,   // 2048 x 512
               const __bf16* __restrict__ Whh,   // 2048 x 512
               const float* __restrict__ bw,     // 2048
               __bf16* __restrict__ out)         // 4096 x 512 (h per step)
{
    __shared__ float xh[1024];   // [x(512) | h(512)]
    __shared__ float zb[2048];

    const int tid = threadIdx.x;
    float c = 0.0f;
    if (tid < 512) xh[512 + tid] = 0.0f;
    __syncthreads();

    const float b0 = bw[tid];
    const float b1 = bw[tid + 1024];
    const __bf16* w0i = Wih + (long)tid * 512;
    const __bf16* w1i = Wih + (long)(tid + 1024) * 512;
    const __bf16* w0h = Whh + (long)tid * 512;
    const __bf16* w1h = Whh + (long)(tid + 1024) * 512;

    for (int t = 0; t < S_; t++) {
        if (tid < 512) xh[tid] = (float)xw[(long)t * 512 + tid];
        if (t + 1 < S_ && tid < 512) __builtin_prefetch(xw + (long)(t + 1) * 512 + tid, 0, 0);
        __syncthreads();

        float a0 = b0, a1 = b1;
        #pragma unroll 4
        for (int kc = 0; kc < 64; kc++) {
            v8bf wa = *(const v8bf*)(w0i + kc * 8);
            v8bf wb = *(const v8bf*)(w1i + kc * 8);
            #pragma unroll
            for (int e = 0; e < 8; e++) {
                float xv = xh[kc * 8 + e];
                a0 += xv * (float)wa[e];
                a1 += xv * (float)wb[e];
            }
        }
        #pragma unroll 4
        for (int kc = 0; kc < 64; kc++) {
            v8bf wa = *(const v8bf*)(w0h + kc * 8);
            v8bf wb = *(const v8bf*)(w1h + kc * 8);
            #pragma unroll
            for (int e = 0; e < 8; e++) {
                float xv = xh[512 + kc * 8 + e];
                a0 += xv * (float)wa[e];
                a1 += xv * (float)wb[e];
            }
        }
        zb[tid] = a0;
        zb[tid + 1024] = a1;
        __syncthreads();

        if (tid < 512) {
            float iv = zb[tid];
            float fv = zb[512 + tid];
            float gv = zb[1024 + tid];
            float ov = zb[1536 + tid];
            c = sigf(fv) * c + sigf(iv) * tanhfast(gv);
            float h = sigf(ov) * tanhfast(c);
            xh[512 + tid] = h;
            out[(long)t * 512 + tid] = (__bf16)h;
        }
        // next-iter top sync orders h writes before next matvec
    }
}

// ---------------- phase 3: tag projection + log_softmax (WMMA) ----------------
__global__ __launch_bounds__(128)
void tag_kernel(const __bf16* __restrict__ A,     // 4096 x 512 (lstm_out)
                const __bf16* __restrict__ Wt,    // 64 x 512
                const float* __restrict__ bt,     // 64
                float* __restrict__ outp)         // 4096 x 64
{
    __shared__ float zt[16 * 64];
    const int tid   = threadIdx.x;
    const int lane  = tid & 31;
    const int wv    = tid >> 5;       // 4 waves -> 4 N-tiles of 16
    const int l15   = lane & 15;
    const int hi    = lane >> 4;
    const int sbase = blockIdx.x * 16;
    const int n     = wv * 16 + l15;

    v8f acc;
    float bv = bt[n];
    #pragma unroll
    for (int r = 0; r < 8; r++) acc[r] = bv;

    const __bf16* arow = A  + (long)(sbase + l15) * 512;
    const __bf16* brow = Wt + (long)n * 512;
    #pragma unroll 2
    for (int kt = 0; kt < 16; kt++) {
        FragB a, b;
        int k0 = kt * 32 + hi * 8;
        a.q[0] = *(const uint4*)(arow + k0);
        a.q[1] = *(const uint4*)(arow + k0 + 16);
        int kb = kt * 32 + hi * 8;
        const uint4* wp = (const uint4*)(brow + kb);
        b.q[0] = wp[0];
        b.q[1] = wp[2];
        acc = __builtin_amdgcn_wmma_f32_16x16x32_bf16(
            false, a.v, false, b.v, (short)0, acc, false, false);
    }
    #pragma unroll
    for (int r = 0; r < 8; r++) {
        int m = hi * 8 + r;
        zt[m * 64 + wv * 16 + l15] = acc[r];
    }
    __syncthreads();

    if (tid < 16) {
        int m = tid;
        float mx = -1e30f;
        for (int cix = 0; cix < T_; cix++) mx = fmaxf(mx, zt[m * 64 + cix]);
        float ssum = 0.0f;
        for (int cix = 0; cix < T_; cix++) ssum += __expf(zt[m * 64 + cix] - mx);
        float lse = mx + __logf(ssum);
        for (int cix = 0; cix < T_; cix++)
            outp[(long)(sbase + m) * 64 + cix] = zt[m * 64 + cix] - lse;
    }
}

// ---------------- launcher ----------------
extern "C" void kernel_launch(void* const* d_in, const int* in_sizes, int n_in,
                              void* d_out, int out_size, void* d_ws, size_t ws_size,
                              hipStream_t stream) {
    const int*   sentence    = (const int*)d_in[0];
    const int*   words_chars = (const int*)d_in[1];
    const int*   word_lens   = (const int*)d_in[2];
    const float* word_emb    = (const float*)d_in[3];
    const float* char_emb    = (const float*)d_in[4];
    const float* Wc_ih       = (const float*)d_in[5];
    const float* Wc_hh       = (const float*)d_in[6];
    const float* bc          = (const float*)d_in[7];
    const float* Ww_ih       = (const float*)d_in[8];
    const float* Ww_hh       = (const float*)d_in[9];
    const float* bw          = (const float*)d_in[10];
    const float* W_tag       = (const float*)d_in[11];
    const float* b_tag       = (const float*)d_in[12];

    char* ws = (char*)d_ws;
    size_t off = 0;
    auto alloc = [&](size_t bytes) -> char* {
        char* p = ws + off;
        off += (bytes + 255) & ~size_t(255);
        return p;
    };
    __bf16* ce_bf   = (__bf16*)alloc(26 * 256 * 2);
    __bf16* Wcih_bf = (__bf16*)alloc(1024 * 256 * 2);
    __bf16* Wchh_bf = (__bf16*)alloc(1024 * 256 * 2);
    __bf16* Wwih_bf = (__bf16*)alloc(2048 * 512 * 2);
    __bf16* Wwhh_bf = (__bf16*)alloc(2048 * 512 * 2);
    __bf16* Wt_bf   = (__bf16*)alloc(64 * 512 * 2);
    __bf16* xw_bf   = (__bf16*)alloc((size_t)S_ * 512 * 2);
    __bf16* lo_bf   = (__bf16*)alloc((size_t)S_ * 512 * 2);

    auto cvt = [&](const float* s, __bf16* d, int n) {
        cvt_bf16<<<(n + 255) / 256, 256, 0, stream>>>(s, d, n);
    };
    cvt(char_emb, ce_bf,   26 * 256);
    cvt(Wc_ih,    Wcih_bf, 1024 * 256);
    cvt(Wc_hh,    Wchh_bf, 1024 * 256);
    cvt(Ww_ih,    Wwih_bf, 2048 * 512);
    cvt(Ww_hh,    Wwhh_bf, 2048 * 512);
    cvt(W_tag,    Wt_bf,   64 * 512);

    gather_we<<<S_, 256, 0, stream>>>(sentence, word_emb, xw_bf);
    char_lstm<<<S_ / 32, 512, 0, stream>>>(words_chars, word_lens, ce_bf,
                                           Wcih_bf, Wchh_bf, bc, xw_bf);
    word_lstm<<<1, 1024, 0, stream>>>(xw_bf, Wwih_bf, Wwhh_bf, bw, lo_bf);
    tag_kernel<<<S_ / 16, 128, 0, stream>>>(lo_bf, Wt_bf, b_tag, (float*)d_out);
}
